// SDEBlock_26972394619157
// MI455X (gfx1250) — compile-verified, matrix operand order checked by reading, not run on hardware
//
#include <hip/hip_runtime.h>
#include <hip/hip_bf16.h>
#include <stdint.h>

// ---------------- problem constants ----------------
#define D_LAT   128
#define HID     256
#define N_SDE   100
#define BATCH   16384
#define HALFB   (BATCH / 2)
#define DT      0.01f
#define SQRT_DT 0.1f

#define WAVES        4
#define ROWS_PER_WG  (WAVES * 16)   // 64 rows per workgroup (32 lower half + 32 upper half)
#define HS           260            // LDS activation row stride (bf16 elems), padded vs 256

typedef __attribute__((ext_vector_type(16))) __bf16 v16bf;
typedef __attribute__((ext_vector_type(8)))  float  v8f;

union Frag { uint32_t u[8]; v16bf v; };

// f32 -> bf16 bits, round-to-nearest-even
static __device__ __forceinline__ unsigned short bfc(float f) {
  uint32_t u = __float_as_uint(f);
  uint32_t r = u + 0x7fffu + ((u >> 16) & 1u);
  return (unsigned short)(r >> 16);
}

static __device__ __forceinline__ uint32_t rotl32(uint32_t x, int r) {
  return (x << r) | (x >> (32 - r));
}

// Threefry-2x32, 20 rounds (JAX PRNG)
static __device__ __forceinline__ void threefry2x32(uint32_t k0, uint32_t k1,
                                                    uint32_t c0, uint32_t c1,
                                                    uint32_t& o0, uint32_t& o1) {
  uint32_t ks2 = k0 ^ k1 ^ 0x1BD11BDAu;
  uint32_t x0 = c0 + k0, x1 = c1 + k1;
  x0 += x1; x1 = rotl32(x1, 13); x1 ^= x0;
  x0 += x1; x1 = rotl32(x1, 15); x1 ^= x0;
  x0 += x1; x1 = rotl32(x1, 26); x1 ^= x0;
  x0 += x1; x1 = rotl32(x1,  6); x1 ^= x0;
  x0 += k1; x1 += ks2 + 1u;
  x0 += x1; x1 = rotl32(x1, 17); x1 ^= x0;
  x0 += x1; x1 = rotl32(x1, 29); x1 ^= x0;
  x0 += x1; x1 = rotl32(x1, 16); x1 ^= x0;
  x0 += x1; x1 = rotl32(x1, 24); x1 ^= x0;
  x0 += ks2; x1 += k0 + 2u;
  x0 += x1; x1 = rotl32(x1, 13); x1 ^= x0;
  x0 += x1; x1 = rotl32(x1, 15); x1 ^= x0;
  x0 += x1; x1 = rotl32(x1, 26); x1 ^= x0;
  x0 += x1; x1 = rotl32(x1,  6); x1 ^= x0;
  x0 += k0; x1 += k1 + 3u;
  x0 += x1; x1 = rotl32(x1, 17); x1 ^= x0;
  x0 += x1; x1 = rotl32(x1, 29); x1 ^= x0;
  x0 += x1; x1 = rotl32(x1, 16); x1 ^= x0;
  x0 += x1; x1 = rotl32(x1, 24); x1 ^= x0;
  x0 += k1; x1 += ks2 + 4u;
  x0 += x1; x1 = rotl32(x1, 13); x1 ^= x0;
  x0 += x1; x1 = rotl32(x1, 15); x1 ^= x0;
  x0 += x1; x1 = rotl32(x1, 26); x1 ^= x0;
  x0 += x1; x1 = rotl32(x1,  6); x1 ^= x0;
  x0 += ks2; x1 += k0 + 5u;
  o0 = x0; o1 = x1;
}

static __device__ __forceinline__ float bits_to_normal(uint32_t bits) {
  float u = __uint_as_float((bits >> 9) | 0x3f800000u) * 2.0f - 3.0f;  // [-1,1)
  return 1.41421356237f * erfinvf(u);
}

// Local row m (0..15) -> global batch row. Rows m&4==0 are lower-half rows,
// m&4==1 upper-half rows, so (m, m+4) form a (e, e+HALF) threefry pair in-lane.
static __device__ __forceinline__ int global_row(int base, int m) {
  return ((m & 4) ? HALFB : 0) + base + ((m >> 3) << 2) + (m & 3);
}

// ---- fragment loads (ISA 7.12.2 layouts) ----
// K index inside a 32-deep chunk for VGPR v, subword s, lane-half h:
//   k = ((v>>2)<<4) + ((v&3)<<1) + s + (h<<3)
static __device__ __forceinline__ Frag load_fragA(const unsigned short* rowp, int kbase, int h) {
  Frag f;
#pragma unroll
  for (int v = 0; v < 8; ++v) {
    int k = kbase + ((v >> 2) << 4) + ((v & 3) << 1) + (h << 3);
    f.u[v] = *(const uint32_t*)(rowp + k);   // s=0 low, s=1 high (contiguous in memory)
  }
  return f;
}

static __device__ __forceinline__ Frag load_fragB(const uint32_t* tile, int lane) {
  Frag f;
  const uint32_t* p = tile + lane * 8;       // pre-swizzled: 8 consecutive dwords per lane
#pragma unroll
  for (int v = 0; v < 8; ++v) f.u[v] = p[v];
  return f;
}

// Convert global f32 weight [K][N] into LDS bf16 B-fragment tiles (32x16 per tile)
static __device__ __forceinline__ void pack_weight(uint32_t* dst, const float* src,
                                                   int K, int N, int tid) {
  const int ntN = N >> 4;
  for (int p = tid; p < (K * N) >> 1; p += 128) {
    int n = p % N;
    int k = (p / N) << 1;                    // even k; pair (k, k+1) share one dword
    uint32_t lo = bfc(src[(size_t)k * N + n]);
    uint32_t hi = bfc(src[(size_t)(k + 1) * N + n]);
    int ck = k >> 5, nt = n >> 4, kk = k & 31;
    int h  = (kk >> 3) & 1;
    int vv = ((kk >> 4) << 2) | ((kk & 7) >> 1);
    int ln = (n & 15) | (h << 4);
    dst[(ck * ntN + nt) * 256 + ln * 8 + vv] = lo | (hi << 16);
  }
}

// One MLP: out = tanh(z @ W1 + b1) @ W2 + b2 for this wave's 16 rows.
// B-fragments flow through a 3-buffer rotation so two tile-loads stay in flight.
static __device__ __forceinline__ void mlp_forward(const Frag zfrag[4],
                                                   const uint32_t* W1, const uint32_t* W2,
                                                   const float* b1, const float* b2,
                                                   unsigned short* hArow,
                                                   int lane, int n16, int hgrp,
                                                   v8f out[8]) {
  const unsigned short* hrowA = hArow + n16 * HS;

  // ---- layer 1: [16x128] @ [128x256], two halves of 8 N-tiles ----
#pragma unroll
  for (int half = 0; half < 2; ++half) {
    v8f acc[8];
#pragma unroll
    for (int t = 0; t < 8; ++t)
#pragma unroll
      for (int r = 0; r < 8; ++r) acc[t][r] = 0.0f;

    // tile index for flat i = c*8 + t  ->  (c*16 + half*8 + t)
#define W1TILE(i) (W1 + ((((i) >> 3) * 16) + half * 8 + ((i) & 7)) * 256)
    Frag bcur = load_fragB(W1TILE(0), lane);
    Frag bnx1 = load_fragB(W1TILE(1), lane);
    Frag bnx2;
#pragma unroll
    for (int i = 0; i < 32; ++i) {
      int c = i >> 3, t = i & 7;
      if (i < 30) bnx2 = load_fragB(W1TILE(i + 2), lane);
      acc[t] = __builtin_amdgcn_wmma_f32_16x16x32_bf16(
          false, zfrag[c].v, false, bcur.v, (short)0, acc[t], false, false);
      bcur = bnx1;
      if (i < 30) bnx1 = bnx2;
    }
#undef W1TILE

    // bias + tanh, stage bf16 hidden activations in LDS (row-major, A-load friendly)
#pragma unroll
    for (int t = 0; t < 8; ++t) {
      int nn = (half * 8 + t) * 16 + n16;
      float bias = b1[nn];
#pragma unroll
      for (int r = 0; r < 8; ++r)
        hArow[(r + 8 * hgrp) * HS + nn] = bfc(tanhf(acc[t][r] + bias));
    }
  }
  asm volatile("s_wait_dscnt 0" ::: "memory");

  // ---- layer 2: [16x256] @ [256x128], W2 tile index == flat i ----
  v8f oacc[8];
#pragma unroll
  for (int t = 0; t < 8; ++t)
#pragma unroll
    for (int r = 0; r < 8; ++r) oacc[t][r] = 0.0f;

  Frag hcur = load_fragA(hrowA, 0, hgrp), hnxt;
  Frag bcur = load_fragB(W2, lane);
  Frag bnx1 = load_fragB(W2 + 256, lane);
  Frag bnx2;
#pragma unroll
  for (int i = 0; i < 64; ++i) {
    int c = i >> 3, t = i & 7;
    if (t == 0 && c < 7) hnxt = load_fragA(hrowA, 32 * (c + 1), hgrp);
    if (i < 62) bnx2 = load_fragB(W2 + (i + 2) * 256, lane);
    oacc[t] = __builtin_amdgcn_wmma_f32_16x16x32_bf16(
        false, hcur.v, false, bcur.v, (short)0, oacc[t], false, false);
    bcur = bnx1;
    if (i < 62) bnx1 = bnx2;
    if (t == 7 && c < 7) hcur = hnxt;
  }

#pragma unroll
  for (int t = 0; t < 8; ++t) {
    float bias = b2[t * 16 + n16];
#pragma unroll
    for (int r = 0; r < 8; ++r) out[t][r] = oacc[t][r] + bias;
  }
}

__global__ __launch_bounds__(128) void sde_kernel(
    const float* __restrict__ z0,
    const float* __restrict__ Wf1, const float* __restrict__ bf1,
    const float* __restrict__ Wf2, const float* __restrict__ bf2,
    const float* __restrict__ Wg1, const float* __restrict__ bg1,
    const float* __restrict__ Wg2, const float* __restrict__ bg2,
    float* __restrict__ zout) {
  // 256 KB of bf16 weights (fragment-swizzled) + biases + per-wave activation staging
  __shared__ uint32_t       sW[4 * 16384];
  __shared__ float          sB[2 * (HID + D_LAT)];
  __shared__ unsigned short sHA[WAVES * 16 * HS];

  const int tid  = threadIdx.x;
  const int lane = tid & 31;
  const int wave = tid >> 5;
  const int n16  = lane & 15;
  const int hgrp = lane >> 4;

  // ---- one-time cooperative weight conversion into LDS ----
  pack_weight(sW +     0, Wf1, D_LAT, HID, tid);
  pack_weight(sW + 16384, Wf2, HID, D_LAT, tid);
  pack_weight(sW + 32768, Wg1, D_LAT, HID, tid);
  pack_weight(sW + 49152, Wg2, HID, D_LAT, tid);
  for (int i = tid; i < 768; i += 128) {
    float v;
    if      (i < 256) v = bf1[i];
    else if (i < 384) v = bf2[i - 256];
    else if (i < 640) v = bg1[i - 384];
    else              v = bg2[i - 640];
    sB[i] = v;
  }
  __syncthreads();

  unsigned short* hArow = sHA + wave * 16 * HS;
  const int base = blockIdx.x * (ROWS_PER_WG / 2) + wave * 8;   // lower-half row base

  // ---- master z state lives in f32 registers (C layout, 8 tiles of 16x16) ----
  v8f zc[8];
#pragma unroll
  for (int t = 0; t < 8; ++t)
#pragma unroll
    for (int r = 0; r < 8; ++r)
      zc[t][r] = z0[(size_t)global_row(base, r + 8 * hgrp) * D_LAT + t * 16 + n16];

#pragma unroll 1
  for (int i = 0; i < N_SDE; ++i) {
    // per-step key: fold_in(key(42), i)
    uint32_t sk0, sk1;
    threefry2x32(0u, 42u, 0u, (uint32_t)i, sk0, sk1);

    // stage z as bf16 in LDS (reuse hidden buffer), then pull A-fragments to regs
#pragma unroll
    for (int t = 0; t < 8; ++t)
#pragma unroll
      for (int r = 0; r < 8; ++r)
        hArow[(r + 8 * hgrp) * HS + t * 16 + n16] = bfc(zc[t][r]);
    asm volatile("s_wait_dscnt 0" ::: "memory");

    Frag zfrag[4];
#pragma unroll
    for (int c = 0; c < 4; ++c)
      zfrag[c] = load_fragA(hArow + n16 * HS, 32 * c, hgrp);
    asm volatile("" ::: "memory");

    // drift MLP; fold into z immediately so fout dies before the g-MLP runs
    v8f fout[8];
    mlp_forward(zfrag, sW + 0, sW + 16384, sB + 0, sB + 256,
                hArow, lane, n16, hgrp, fout);
#pragma unroll
    for (int t = 0; t < 8; ++t)
#pragma unroll
      for (int r = 0; r < 8; ++r) zc[t][r] += fout[t][r] * DT;

    // diffusion MLP
    v8f gout[8];
    mlp_forward(zfrag, sW + 32768, sW + 49152, sB + 384, sB + 640,
                hArow, lane, n16, hgrp, gout);

    // noise: rows (m, m+4) are (e, e+HALF) partners -> one threefry per pair
#pragma unroll
    for (int t = 0; t < 8; ++t)
#pragma unroll
      for (int r = 0; r < 4; ++r) {
        int m = r + 8 * hgrp;                       // partner is m+4 (VGPR r+4)
        uint32_t e = (uint32_t)global_row(base, m) * (uint32_t)D_LAT
                   + (uint32_t)(t * 16 + n16);      // e < HALFB*D_LAT by construction
        uint32_t o0, o1;
        threefry2x32(sk0, sk1, e, e + (uint32_t)(HALFB * D_LAT), o0, o1);
        zc[t][r]     += gout[t][r]     * (SQRT_DT * bits_to_normal(o0));
        zc[t][r + 4] += gout[t][r + 4] * (SQRT_DT * bits_to_normal(o1));
      }
  }

#pragma unroll
  for (int t = 0; t < 8; ++t)
#pragma unroll
    for (int r = 0; r < 8; ++r)
      zout[(size_t)global_row(base, r + 8 * hgrp) * D_LAT + t * 16 + n16] = zc[t][r];
}

extern "C" void kernel_launch(void* const* d_in, const int* in_sizes, int n_in,
                              void* d_out, int out_size, void* d_ws, size_t ws_size,
                              hipStream_t stream) {
  (void)in_sizes; (void)n_in; (void)d_ws; (void)ws_size; (void)out_size;
  const float* z0  = (const float*)d_in[0];
  const float* Wf1 = (const float*)d_in[1];
  const float* bf1 = (const float*)d_in[2];
  const float* Wf2 = (const float*)d_in[3];
  const float* bf2 = (const float*)d_in[4];
  const float* Wg1 = (const float*)d_in[5];
  const float* bg1 = (const float*)d_in[6];
  const float* Wg2 = (const float*)d_in[7];
  const float* bg2 = (const float*)d_in[8];
  float* out = (float*)d_out;

  dim3 grid(BATCH / ROWS_PER_WG);   // 256 workgroups -> one per WGP
  dim3 block(128);                  // 4 wave32 waves
  sde_kernel<<<grid, block, 0, stream>>>(z0, Wf1, bf1, Wf2, bf2, Wg1, bg1, Wg2, bg2, out);
}
